// GraphExtractor_33930241638445
// MI455X (gfx1250) — matrix operand order repaired and checked by reading, not compiled, not run on hardware
//
#include <hip/hip_runtime.h>

// ---------------- problem constants ----------------
#define BB   8
#define SS   4096
#define HH   768
#define NHH  12
#define LL   2
#define NSS  64
#define DHH  64
#define TOK  64
#define ROWS (BB*NSS)        // 512

typedef __attribute__((ext_vector_type(16))) __bf16 v16bf;
typedef __attribute__((ext_vector_type(8)))  float  v8f;
typedef __attribute__((ext_vector_type(4)))  unsigned int u32x4;
typedef __attribute__((ext_vector_type(4)))  int v4i;
typedef unsigned short u16;
typedef unsigned int   u32;

union Frag { u32 u[8]; v16bf v; };

__device__ __forceinline__ u16 f2bf(float f) {
  u32 u = __builtin_bit_cast(u32, f);
  u32 r = u + 0x7FFFu + ((u >> 16) & 1u);      // RNE
  return (u16)(r >> 16);
}

// -------- gfx1250 async global->LDS copy (guarded; falls back to b128 ld/st) --------
#if defined(__has_builtin)
#if __has_builtin(__builtin_amdgcn_global_load_async_to_lds_b128) && \
    __has_builtin(__builtin_amdgcn_s_wait_asynccnt)
#define HAVE_ASYNC 1
#endif
#endif

__device__ __forceinline__ void copy16(const u16* g, u16* l) {
#ifdef HAVE_ASYNC
  __builtin_amdgcn_global_load_async_to_lds_b128(
      (__attribute__((address_space(1))) v4i*)(void*)(size_t)(const void*)g,
      (__attribute__((address_space(3))) v4i*)(void*)l, 0, 0);
#else
  *(u32x4*)l = *(const u32x4*)g;
#endif
}
__device__ __forceinline__ void async_join() {
#ifdef HAVE_ASYNC
  __builtin_amdgcn_s_wait_asynccnt(0);
#endif
}

// ---------------- 0) weight convert + transpose: Wt[l][n][k] bf16 ----------------
// grid (12, 12, 8=mat*2+layer), block 256
__global__ __launch_bounds__(256) void k_cvtw(const float* __restrict__ Wq,
                                              const float* __restrict__ Wk,
                                              const float* __restrict__ Wv,
                                              const float* __restrict__ Wo,
                                              u16* __restrict__ dst) {
  __shared__ __attribute__((aligned(16))) u16 t[64 * 64];
  const int z = blockIdx.z, mat = z >> 1, li = z & 1;
  const float* src = (mat == 0 ? Wq : mat == 1 ? Wk : mat == 2 ? Wv : Wo) + (size_t)li * HH * HH;
  u16* out = dst + (size_t)z * HH * HH;
  const int kb = blockIdx.y * 64, nb = blockIdx.x * 64;
  const int tid = threadIdx.x;
#pragma unroll
  for (int e = 0; e < 16; ++e) {
    int idx = e * 256 + tid;             // 0..4095
    int r = idx >> 6, c = idx & 63;      // r = k-local, c = n-local (coalesced read)
    t[c * 64 + r] = f2bf(src[(size_t)(kb + r) * HH + nb + c]);
  }
  __syncthreads();
#pragma unroll
  for (int e = 0; e < 8; ++e) {
    int idx = e * 256 + tid;             // 0..2047 (u32 pairs)
    int n = idx >> 5, kp = idx & 31;
    *(u32*)&out[(size_t)(nb + n) * HH + kb + 2 * kp] = *(const u32*)&t[n * 64 + 2 * kp];
  }
}

// ---------------- 1) segment-mean pooling -> bf16 x ----------------
__global__ __launch_bounds__(192) void k_pool(const float* __restrict__ SL,
                                              u16* __restrict__ X) {
  const int blk = blockIdx.x, b = blk >> 6, n = blk & 63;
  const float* base = SL + ((size_t)b * SS + (size_t)n * TOK) * HH;
  const int t0 = (n == 0) ? 1 : 0;
  const float invc = (n == 0) ? (1.0f / 63.0f) : (1.0f / 64.0f);
  const int d4 = threadIdx.x;            // 0..191, one float4 column each
  const float4* p = (const float4*)(base + d4 * 4);
  float s0 = 0.f, s1 = 0.f, s2 = 0.f, s3 = 0.f;
  for (int t = t0; t < TOK; ++t) {
    float4 v = p[t * (HH / 4)];
    s0 += v.x; s1 += v.y; s2 += v.z; s3 += v.w;
  }
  uint2 r;
  r.x = (u32)f2bf(s0 * invc) | ((u32)f2bf(s1 * invc) << 16);
  r.y = (u32)f2bf(s2 * invc) | ((u32)f2bf(s3 * invc) << 16);
  *(uint2*)&X[((size_t)b * NSS + n) * HH + d4 * 4] = r;
}

// ---------------- 2) GEMM: [512,768] x Wt(bf16,[n][k]) + bias ----------------
// grid (12, 8, nz), block 128; MODE 0: bf16 out, MODE 1: f32 out + exact GELU
template <int MODE>
__global__ __launch_bounds__(128) void k_gemm(const u16* __restrict__ A,
                                              const u16* __restrict__ Wt, long wstride,
                                              const float* __restrict__ b0,
                                              const float* __restrict__ b1,
                                              const float* __restrict__ b2,
                                              void* o0, void* o1, void* o2) {
  __shared__ __attribute__((aligned(16))) u16 As[64 * 32];
  __shared__ __attribute__((aligned(16))) u16 Bs[64 * 32];

  const int z = blockIdx.z;
  const u16* W = Wt + (size_t)z * (size_t)wstride;
  const float* bias = (z == 0) ? b0 : (z == 1) ? b1 : b2;
  void* op = (z == 0) ? o0 : (z == 1) ? o1 : o2;

  const int tid = threadIdx.x;
  const int tileN = blockIdx.x * 64, tileM = blockIdx.y * 64;
  const int wave = tid >> 5, lane = tid & 31, ln15 = lane & 15, hi = lane >> 4;
  const int wm = (wave >> 1) * 32, wn = (wave & 1) * 32;

  v8f c[2][2];
  c[0][0] = {}; c[0][1] = {}; c[1][0] = {}; c[1][1] = {};

  for (int kt = 0; kt < HH; kt += 32) {
    // both tiles are 64 rows x 32 bf16 (64B) = 256 chunks of 16B each
#pragma unroll
    for (int e = 0; e < 2; ++e) {
      int ch = e * 128 + tid;            // 0..255
      int row = ch >> 2, part = (ch & 3) * 8;
      copy16(&A[(size_t)(tileM + row) * HH + kt + part], &As[row * 32 + part]);
      copy16(&W[(size_t)(tileN + row) * HH + kt + part], &Bs[row * 32 + part]);
    }
    if (kt + 32 < HH)
      __builtin_prefetch(&W[(size_t)(tileN + (tid >> 1)) * HH + kt + 32], 0, 1);
    async_join();
    __syncthreads();

    Frag a[2], bfr[2];
#pragma unroll
    for (int mi = 0; mi < 2; ++mi) {
      const u16* rp = &As[(wm + mi * 16 + ln15) * 32];
#pragma unroll
      for (int i = 0; i < 8; ++i) {
        int k0 = (i < 4) ? (hi * 8 + 2 * i) : (16 + hi * 8 + 2 * (i - 4));
        a[mi].u[i] = *(const u32*)(rp + k0);
      }
    }
#pragma unroll
    for (int ni = 0; ni < 2; ++ni) {
      const u16* rp = &Bs[(wn + ni * 16 + ln15) * 32];
#pragma unroll
      for (int i = 0; i < 8; ++i)
        bfr[ni].u[i] = *(const u32*)(rp + ((i < 4) ? (hi * 8 + 2 * i) : (16 + hi * 8 + 2 * (i - 4))));
    }
#pragma unroll
    for (int mi = 0; mi < 2; ++mi)
#pragma unroll
      for (int ni = 0; ni < 2; ++ni)
        c[mi][ni] = __builtin_amdgcn_wmma_f32_16x16x32_bf16(
            false, a[mi].v, false, bfr[ni].v, (short)0, c[mi][ni], false, false);
    __syncthreads();
  }

#pragma unroll
  for (int mi = 0; mi < 2; ++mi) {
#pragma unroll
    for (int ni = 0; ni < 2; ++ni) {
      const int ng = tileN + wn + ni * 16 + ln15;
      const float bv = bias[ng];
#pragma unroll
      for (int r = 0; r < 8; ++r) {
        const int mg = tileM + wm + mi * 16 + hi * 8 + r;
        float v = c[mi][ni][r] + bv;
        if (MODE == 1) {
          v = 0.5f * v * (1.0f + erff(v * 0.70710678118654752f));
          ((float*)op)[(size_t)mg * HH + ng] = v;
        } else {
          ((u16*)op)[(size_t)mg * HH + ng] = f2bf(v);
        }
      }
    }
  }
}

// ---------------- 3) attention per (b, head) ----------------
__global__ __launch_bounds__(128) void k_attn(const u16* __restrict__ Q,
                                              const u16* __restrict__ Kb,
                                              const u16* __restrict__ V,
                                              u16* __restrict__ Ct) {
  __shared__ __attribute__((aligned(16))) u16 q_s[64 * 64];
  __shared__ __attribute__((aligned(16))) u16 k_s[64 * 64];
  __shared__ __attribute__((aligned(16))) u16 vT_s[64 * 64];
  __shared__ __attribute__((aligned(16))) u16 p_s[64 * 64];

  const int b = blockIdx.x / NHH, h = blockIdx.x % NHH;
  const int tid = threadIdx.x;
  const int wave = tid >> 5, lane = tid & 31, ln15 = lane & 15, hi = lane >> 4;
  const size_t rowBase = (size_t)b * NSS * HH + (size_t)h * DHH;

  // q, k: straight 16B-chunk copies (64 rows x 128B = 512 chunks each)
#pragma unroll
  for (int e = 0; e < 4; ++e) {
    int ch = e * 128 + tid;
    int m = ch >> 3, part = (ch & 7) * 8;
    copy16(&Q[rowBase + (size_t)m * HH + part], &q_s[m * 64 + part]);
    copy16(&Kb[rowBase + (size_t)m * HH + part], &k_s[m * 64 + part]);
  }
  // v: transpose into [dh][ns]
#pragma unroll
  for (int e = 0; e < 32; ++e) {
    int idx = e * 128 + tid;
    int m = idx >> 6, d = idx & 63;
    vT_s[d * 64 + m] = V[rowBase + (size_t)m * HH + d];
  }
  async_join();
  __syncthreads();

  const int m0 = wave * 16;

  // scores = q @ k^T
  v8f s[4]; s[0] = {}; s[1] = {}; s[2] = {}; s[3] = {};
#pragma unroll
  for (int kk = 0; kk < DHH; kk += 32) {
    Frag a;
    const u16* arp = &q_s[(m0 + ln15) * 64 + kk];
#pragma unroll
    for (int i = 0; i < 8; ++i) {
      int k0 = (i < 4) ? (hi * 8 + 2 * i) : (16 + hi * 8 + 2 * (i - 4));
      a.u[i] = *(const u32*)(arp + k0);
    }
#pragma unroll
    for (int nt = 0; nt < 4; ++nt) {
      Frag bf;
      const u16* brp = &k_s[(nt * 16 + ln15) * 64 + kk];
#pragma unroll
      for (int i = 0; i < 8; ++i)
        bf.u[i] = *(const u32*)(brp + ((i < 4) ? (hi * 8 + 2 * i) : (16 + hi * 8 + 2 * (i - 4))));
      s[nt] = __builtin_amdgcn_wmma_f32_16x16x32_bf16(
          false, a.v, false, bf.v, (short)0, s[nt], false, false);
    }
  }

  // softmax (rows live in r + 16-lane half; reductions stay inside the half)
#pragma unroll
  for (int nt = 0; nt < 4; ++nt)
#pragma unroll
    for (int r = 0; r < 8; ++r) s[nt][r] *= 0.125f;

#pragma unroll
  for (int r = 0; r < 8; ++r) {
    float mx = fmaxf(fmaxf(s[0][r], s[1][r]), fmaxf(s[2][r], s[3][r]));
    for (int off = 1; off < 16; off <<= 1) mx = fmaxf(mx, __shfl_xor(mx, off));
    float e0[4], sum = 0.0f;
#pragma unroll
    for (int nt = 0; nt < 4; ++nt) { e0[nt] = __expf(s[nt][r] - mx); sum += e0[nt]; }
    for (int off = 1; off < 16; off <<= 1) sum += __shfl_xor(sum, off);
    const float rinv = 1.0f / sum;
    const int m = m0 + hi * 8 + r;
#pragma unroll
    for (int nt = 0; nt < 4; ++nt)
      p_s[m * 64 + nt * 16 + ln15] = f2bf(e0[nt] * rinv);
  }
  __syncthreads();

  // ctx = probs @ v
  v8f o[4]; o[0] = {}; o[1] = {}; o[2] = {}; o[3] = {};
#pragma unroll
  for (int kk = 0; kk < NSS; kk += 32) {
    Frag a;
    const u16* arp = &p_s[(m0 + ln15) * 64 + kk];
#pragma unroll
    for (int i = 0; i < 8; ++i) {
      int k0 = (i < 4) ? (hi * 8 + 2 * i) : (16 + hi * 8 + 2 * (i - 4));
      a.u[i] = *(const u32*)(arp + k0);
    }
#pragma unroll
    for (int dt = 0; dt < 4; ++dt) {
      Frag bf;
      const u16* brp = &vT_s[(dt * 16 + ln15) * 64 + kk];
#pragma unroll
      for (int i = 0; i < 8; ++i)
        bf.u[i] = *(const u32*)(brp + ((i < 4) ? (hi * 8 + 2 * i) : (16 + hi * 8 + 2 * (i - 4))));
      o[dt] = __builtin_amdgcn_wmma_f32_16x16x32_bf16(
          false, a.v, false, bf.v, (short)0, o[dt], false, false);
    }
  }
#pragma unroll
  for (int dt = 0; dt < 4; ++dt) {
    const int d = dt * 16 + ln15;
#pragma unroll
    for (int r = 0; r < 8; ++r) {
      const int m = m0 + hi * 8 + r;
      Ct[rowBase + (size_t)m * HH + d] = f2bf(o[dt][r]);
    }
  }
}

// ---------------- 4) LayerNorm; OUT16=1 -> bf16 x, OUT16=0 -> f32 out ----------------
template <int OUT16>
__global__ __launch_bounds__(256) void k_ln(const float* __restrict__ Hb,
                                            const float* __restrict__ g,
                                            const float* __restrict__ be,
                                            void* __restrict__ Xo) {
  __shared__ float red0[8], red1[8];
  const int row = blockIdx.x;
  const float* hr = Hb + (size_t)row * HH;
  float v0[3], s = 0.0f, s2 = 0.0f;
#pragma unroll
  for (int i = 0; i < 3; ++i) {
    float v = hr[threadIdx.x + i * 256];
    v0[i] = v; s += v; s2 += v * v;
  }
  for (int off = 1; off < 32; off <<= 1) { s += __shfl_xor(s, off); s2 += __shfl_xor(s2, off); }
  const int lane = threadIdx.x & 31, wv = threadIdx.x >> 5;
  if (lane == 0) { red0[wv] = s; red1[wv] = s2; }
  __syncthreads();
  if (wv == 0) {
    float a = (lane < 8) ? red0[lane] : 0.0f;
    float b = (lane < 8) ? red1[lane] : 0.0f;
    for (int off = 1; off < 8; off <<= 1) { a += __shfl_xor(a, off); b += __shfl_xor(b, off); }
    if (lane == 0) { red0[0] = a; red1[0] = b; }
  }
  __syncthreads();
  const float mean = red0[0] * (1.0f / HH);
  const float var  = red1[0] * (1.0f / HH) - mean * mean;
  const float rstd = rsqrtf(var + 1e-12f);
#pragma unroll
  for (int i = 0; i < 3; ++i) {
    const int d = threadIdx.x + i * 256;
    float v = (v0[i] - mean) * rstd * g[d] + be[d];
    if (OUT16) ((u16*)Xo)[(size_t)row * HH + d] = f2bf(v);
    else       ((float*)Xo)[(size_t)row * HH + d] = v;
  }
}

// ---------------- host entry ----------------
extern "C" void kernel_launch(void* const* d_in, const int* in_sizes, int n_in,
                              void* d_out, int out_size, void* d_ws, size_t ws_size,
                              hipStream_t stream) {
  const float* SL = (const float*)d_in[1];
  const float* Wq = (const float*)d_in[2];
  const float* bq = (const float*)d_in[3];
  const float* Wk = (const float*)d_in[4];
  const float* bk = (const float*)d_in[5];
  const float* Wv = (const float*)d_in[6];
  const float* bv = (const float*)d_in[7];
  const float* Wo = (const float*)d_in[8];
  const float* bo = (const float*)d_in[9];
  const float* lg = (const float*)d_in[10];
  const float* lb = (const float*)d_in[11];

  const size_t N = (size_t)ROWS * HH;            // 512*768
  const size_t WE = (size_t)HH * HH;             // 589824
  u16* wt = (u16*)d_ws;                          // 8 * WE bf16 (index = mat*2+layer)
  u16* xb = wt + 8 * WE;
  u16* qb = xb + N;
  u16* kb = qb + N;
  u16* vb = kb + N;
  u16* cb = vb + N;
  float* hb = (float*)(cb + N);
  float* out = (float*)d_out;

  k_cvtw<<<dim3(12, 12, 8), dim3(256), 0, stream>>>(Wq, Wk, Wv, Wo, wt);
  k_pool<<<dim3(ROWS), dim3(192), 0, stream>>>(SL, xb);

  for (int i = 0; i < LL; ++i) {
    const size_t bOff = (size_t)i * HH;
    // Q,K,V in one launch: z selects mat; weights at wt[(mat*2+i)*WE]
    k_gemm<0><<<dim3(12, 8, 3), dim3(128), 0, stream>>>(
        xb, wt + (size_t)i * WE, (long)(2 * WE),
        bq + bOff, bk + bOff, bv + bOff, (void*)qb, (void*)kb, (void*)vb);
    k_attn<<<dim3(BB * NHH), dim3(128), 0, stream>>>(qb, kb, vb, cb);
    k_gemm<1><<<dim3(12, 8, 1), dim3(128), 0, stream>>>(
        cb, wt + (size_t)(3 * 2 + i) * WE, 0L,
        bo + bOff, bo + bOff, bo + bOff, (void*)hb, (void*)hb, (void*)hb);
    if (i == LL - 1)
      k_ln<0><<<dim3(ROWS), dim3(256), 0, stream>>>(hb, lg + bOff, lb + bOff, (void*)out);
    else
      k_ln<1><<<dim3(ROWS), dim3(256), 0, stream>>>(hb, lg + bOff, lb + bOff, (void*)xb);
  }
}
